// XCA_28870770163746
// MI455X (gfx1250) — matrix operand order repaired and checked by reading, not compiled, not run on hardware
//
#include <hip/hip_runtime.h>
#include <hip/hip_bf16.h>

typedef __attribute__((ext_vector_type(16))) __bf16 v16bf;
typedef __attribute__((ext_vector_type(8)))  float  v8f;

#define DIMC 768
#define NTOK 4096
#define NBAT 8
#define NH   8
#define DH   96
#define BH   (NBAT * NH)     // 64
#define MTOT (NBAT * NTOK)   // 32768

// ---------------- WMMA fragment helpers (CDNA5 16x16x32 bf16) ----------------

// A matrix 16x32 (MxK), source row-major: element (m,k) = base[m*ld + k]
// Lanes 0-15: row M=lane, halves = K {0..7, 16..23};
// lanes 16-31: same rows, K {8..15, 24..31}.  -> two contiguous 16B reads.
__device__ inline v16bf ld_frag_a_rm(const __bf16* base, int ld) {
  const int lane = threadIdx.x & 31;
  const __bf16* p = base + (lane & 15) * ld + ((lane >> 4) << 3);
  v16bf f;
#pragma unroll
  for (int i = 0; i < 8; ++i) { f[i] = p[i]; f[i + 8] = p[16 + i]; }
  return f;
}

// B matrix 32x16 (KxN), source k-major: element (k,n) = base[k*ld + n]
// Lane holds column N=lane&15; lanes 0-15: K {0..7, 16..23},
// lanes 16-31: K {8..15, 24..31} (same k order as A).
__device__ inline v16bf ld_frag_b_km(const __bf16* base, int ld) {
  const int lane = threadIdx.x & 31;
  const __bf16* p = base + (lane & 15) + ((lane >> 4) << 3) * ld;
  v16bf f;
#pragma unroll
  for (int c = 0; c < 8; ++c) { f[c] = p[c * ld]; f[c + 8] = p[(16 + c) * ld]; }
  return f;
}

__device__ inline v8f wmma_bf16(v16bf a, v16bf b, v8f c) {
  return __builtin_amdgcn_wmma_f32_16x16x32_bf16(false, a, false, b, (short)0, c,
                                                 false, false);
}

// Pack 8 fp32 accumulators -> 8 bf16 and store as one 16-byte write.
__device__ inline void store_bf16x8(__bf16* p, const v8f& a) {
  union { __bf16 h[8]; uint4 u; } pk;
#pragma unroll
  for (int i = 0; i < 8; ++i) pk.h[i] = (__bf16)a[i];
  *(uint4*)p = pk.u;
}

// ---------------- Kernel 1: weights fp32 -> bf16, transposed to k-major -----
__global__ __launch_bounds__(256) void cvt_weights_kernel(
    const float* __restrict__ qkvw, const float* __restrict__ projw,
    __bf16* __restrict__ WqT, __bf16* __restrict__ WpT) {
  const int stride = gridDim.x * blockDim.x;
  const int t = blockIdx.x * blockDim.x + threadIdx.x;
  const int n1 = 3 * DIMC * DIMC;
  const int n2 = DIMC * DIMC;
  for (int j = t; j < n1; j += stride) {
    const int k = j / (3 * DIMC), n = j - k * (3 * DIMC);
    WqT[j] = (__bf16)qkvw[(size_t)n * DIMC + k];
  }
  for (int j = t; j < n2; j += stride) {
    const int k = j / DIMC, n = j - k * DIMC;
    WpT[j] = (__bf16)projw[(size_t)n * DIMC + k];
  }
}

// ---------------- Kernel 2: LayerNorm -> bf16 ----------------
__global__ __launch_bounds__(256) void ln_kernel(
    const float* __restrict__ X, const float* __restrict__ g,
    const float* __restrict__ b, __bf16* __restrict__ Xn) {
  __shared__ float r1[256], r2[256];
  const int row = blockIdx.x;
  const int tid = threadIdx.x;
  const float* xr = X + (size_t)row * DIMC;
  float v0 = xr[tid], v1 = xr[tid + 256], v2 = xr[tid + 512];
  r1[tid] = v0 + v1 + v2;
  r2[tid] = v0 * v0 + v1 * v1 + v2 * v2;
  __syncthreads();
  for (int o = 128; o > 0; o >>= 1) {
    if (tid < o) { r1[tid] += r1[tid + o]; r2[tid] += r2[tid + o]; }
    __syncthreads();
  }
  const float mu = r1[0] * (1.0f / DIMC);
  const float var = fmaxf(r2[0] * (1.0f / DIMC) - mu * mu, 0.0f);
  const float rstd = rsqrtf(var + 1e-5f);
  __bf16* xo = Xn + (size_t)row * DIMC;
  xo[tid]       = (__bf16)((v0 - mu) * rstd * g[tid]       + b[tid]);
  xo[tid + 256] = (__bf16)((v1 - mu) * rstd * g[tid + 256] + b[tid + 256]);
  xo[tid + 512] = (__bf16)((v2 - mu) * rstd * g[tid + 512] + b[tid + 512]);
}

// ---------------- Kernel 3: QKV GEMM ----------------
// Block 256(M) x 128(N), 8 waves as 4x2 grid of 64x64 tiles, 16 WMMA/k-step.
// Direct global->LDS staging (spill-free); latency hidden by occupancy.
__global__ __launch_bounds__(256) void qkv_gemm_kernel(
    const __bf16* __restrict__ Xn, const __bf16* __restrict__ WqT,
    __bf16* __restrict__ qt, __bf16* __restrict__ kt2, __bf16* __restrict__ vt) {
  __shared__ __attribute__((aligned(16))) __bf16 As[256][40];
  __shared__ __attribute__((aligned(16))) __bf16 Bs[32][136];
  const int tid = threadIdx.x;
  const int lane = tid & 31, wv = tid >> 5;
  const int m0 = blockIdx.y * 256;   // tokens
  const int n0 = blockIdx.x * 128;   // output channels
  const int wm = wv >> 1, wn = wv & 1;

  v8f acc[4][4] = {};
  for (int k0 = 0; k0 < DIMC; k0 += 32) {
#pragma unroll
    for (int it = 0; it < 4; ++it) {
      const int g = tid + it * 256;           // 256 rows x 4 segs
      const int r = g >> 2, seg = g & 3;
      *(uint4*)&As[r][seg * 8] =
          *(const uint4*)&Xn[(size_t)(m0 + r) * DIMC + k0 + seg * 8];
    }
#pragma unroll
    for (int it = 0; it < 2; ++it) {
      const int g = tid + it * 256;           // 32 rows x 16 segs
      const int r = g >> 4, seg = g & 15;
      *(uint4*)&Bs[r][seg * 8] =
          *(const uint4*)&WqT[(size_t)(k0 + r) * (3 * DIMC) + n0 + seg * 8];
    }
    __syncthreads();
    v16bf af[4], bfv[4];
#pragma unroll
    for (int i = 0; i < 4; ++i) af[i] = ld_frag_a_rm(&As[wm * 64 + i * 16][0], 40);
#pragma unroll
    for (int j = 0; j < 4; ++j) bfv[j] = ld_frag_b_km(&Bs[0][wn * 64 + j * 16], 136);
#pragma unroll
    for (int i = 0; i < 4; ++i)
#pragma unroll
      for (int j = 0; j < 4; ++j) acc[i][j] = wmma_bf16(af[i], bfv[j], acc[i][j]);
    __syncthreads();
  }

  const int hi = lane >> 4, col = lane & 15;
  const int s = n0 / DIMC;                    // uniform across the block
  __bf16* const base_sv = (s == 0) ? qt : vt; // used when s != 1
  const int nblk = n0 - s * DIMC;
#pragma unroll
  for (int j = 0; j < 4; ++j) {
    const int co = nblk + wn * 64 + j * 16 + col;   // 0..767 within segment
    const int h = co / DH, d = co - h * DH;
#pragma unroll
    for (int i = 0; i < 4; ++i) {
      const int tm = m0 + wm * 64 + i * 16;
      const int b = tm >> 12;
      const int ntok = (tm & (NTOK - 1)) + hi * 8;
      if (s == 1) {
        // k: token-major [b,h][n][d]
        __bf16* dst = kt2 + ((size_t)(b * NH + h) * NTOK + ntok) * DH + d;
#pragma unroll
        for (int r = 0; r < 8; ++r) dst[(size_t)r * DH] = (__bf16)acc[i][j][r];
      } else {
        __bf16* dst = base_sv + ((size_t)((b * NH + h) * DH + d) << 12) + ntok;
        store_bf16x8(dst, acc[i][j]);
      }
    }
  }
}

// ---------------- Kernel 4a: q L2 norms (channel-major rows) ----------------
__global__ __launch_bounds__(256) void qnorm_kernel(
    const __bf16* __restrict__ qt, float* __restrict__ qn) {
  __shared__ float red[256];
  const int row = blockIdx.x;
  const __bf16* src = qt + (size_t)row * NTOK;
  float ss = 0.0f;
  for (int e = threadIdx.x; e < NTOK; e += 256) {
    const float v = (float)src[e];
    ss += v * v;
  }
  red[threadIdx.x] = ss;
  __syncthreads();
  for (int o = 128; o > 0; o >>= 1) {
    if (threadIdx.x < o) red[threadIdx.x] += red[threadIdx.x + o];
    __syncthreads();
  }
  if (threadIdx.x == 0) qn[row] = sqrtf(red[0]);
}

// ---------------- Kernel 4b: k L2 norms (token-major layout) ----------------
__global__ __launch_bounds__(128) void knorm_kernel(
    const __bf16* __restrict__ kt2, float* __restrict__ kn) {
  const int bh = blockIdx.x;
  const int d = threadIdx.x;
  if (d >= DH) return;
  const __bf16* base = kt2 + (size_t)bh * NTOK * DH + d;
  float s0 = 0.f, s1 = 0.f, s2 = 0.f, s3 = 0.f;
#pragma unroll 1
  for (int n = 0; n < NTOK; n += 4) {
    const float a = (float)base[(size_t)(n + 0) * DH];
    const float b = (float)base[(size_t)(n + 1) * DH];
    const float c = (float)base[(size_t)(n + 2) * DH];
    const float e = (float)base[(size_t)(n + 3) * DH];
    s0 += a * a; s1 += b * b; s2 += c * c; s3 += e * e;
  }
  kn[bh * DH + d] = sqrtf((s0 + s1) + (s2 + s3));
}

// ---------------- Kernel 5: S = q k^T (K=4096) + scaled softmax ----------------
// 192 threads = 6 waves; k-step 64 -> 12 WMMA per barrier pair, register
// prefetch (low pressure here: 48 acc VGPRs). Softmax scratch aliases LDS.
__global__ __launch_bounds__(192) void attn_kernel(
    const __bf16* __restrict__ qt, const __bf16* __restrict__ kt2,
    const float* __restrict__ qn, const float* __restrict__ kn,
    const float* __restrict__ temperature, __bf16* __restrict__ attnb) {
  // Qs[96][72] (13824 B) + Ks[64][104] (13312 B) aliased with Sb[96][100] f32
  __shared__ __attribute__((aligned(16))) char smem[DH * 100 * 4];  // 38400 B
  __shared__ float iK[DH];
  __bf16 (*Qs)[72]  = (__bf16(*)[72])smem;                   // [d][k-token]
  __bf16 (*Ks)[104] = (__bf16(*)[104])(smem + DH * 72 * 2);  // [k-token][e]
  float  (*Sb)[100] = (float(*)[100])smem;

  const int bh = blockIdx.x;
  const int tid = threadIdx.x, lane = tid & 31, wv = tid >> 5;
  const __bf16* qb = qt + (size_t)bh * DH * NTOK;
  const __bf16* kb = kt2 + (size_t)bh * NTOK * DH;

  uint4 qa[4], ka[4];
  auto load_qk = [&](int k0) {
#pragma unroll
    for (int it = 0; it < 4; ++it) {
      const int g = tid + it * 192;
      {
        const int r = g >> 3, seg = g & 7;               // 96 rows x 8 segs
        qa[it] = *(const uint4*)&qb[(size_t)r * NTOK + k0 + seg * 8];
      }
      {
        const int r = g / 12, seg = g - (g / 12) * 12;   // 64 rows x 12 segs
        ka[it] = *(const uint4*)&kb[(size_t)(k0 + r) * DH + seg * 8];
      }
    }
  };
  auto store_qk = [&]() {
#pragma unroll
    for (int it = 0; it < 4; ++it) {
      const int g = tid + it * 192;
      {
        const int r = g >> 3, seg = g & 7;
        *(uint4*)&Qs[r][seg * 8] = qa[it];
      }
      {
        const int r = g / 12, seg = g - (g / 12) * 12;
        *(uint4*)&Ks[r][seg * 8] = ka[it];
      }
    }
  };

  v8f acc[6] = {};
  load_qk(0);
  for (int k0 = 0; k0 < NTOK; k0 += 64) {
    store_qk();
    __syncthreads();
    if (k0 + 64 < NTOK) load_qk(k0 + 64);
#pragma unroll
    for (int kk = 0; kk < 2; ++kk) {
      const v16bf a = ld_frag_a_rm(&Qs[wv * 16][kk * 32], 72);
#pragma unroll
      for (int j = 0; j < 6; ++j) {
        const v16bf bfr = ld_frag_b_km(&Ks[kk * 32][j * 16], 104);
        acc[j] = wmma_bf16(a, bfr, acc[j]);
      }
    }
    __syncthreads();
  }

  // GEMM staging is dead now; reuse the LDS as the fp32 score matrix.
  {
    const int col = lane & 15, hi = lane >> 4;
#pragma unroll
    for (int j = 0; j < 6; ++j)
#pragma unroll
      for (int r = 0; r < 8; ++r)
        Sb[wv * 16 + hi * 8 + r][j * 16 + col] = acc[j][r];
  }
  if (tid < DH) iK[tid] = 1.0f / fmaxf(kn[bh * DH + tid], 1e-12f);
  __syncthreads();

  if (tid < DH) {
    const int d = tid;
    const int h = bh & (NH - 1);
    const float sc = (1.0f / fmaxf(qn[bh * DH + d], 1e-12f)) * temperature[h];
    float mx = -3.4e38f;
    for (int e = 0; e < DH; ++e) {
      const float v = Sb[d][e] * sc * iK[e];
      Sb[d][e] = v;
      mx = fmaxf(mx, v);
    }
    float sum = 0.0f;
    for (int e = 0; e < DH; ++e) {
      const float p = __expf(Sb[d][e] - mx);
      Sb[d][e] = p;
      sum += p;
    }
    const float inv = 1.0f / sum;
    __bf16* dst = attnb + (size_t)bh * DH * DH + (size_t)d * DH;
    for (int e = 0; e < DH; ++e) dst[e] = (__bf16)(Sb[d][e] * inv);
  }
}

// ---------------- Kernel 6: Y = attn @ v, store to [B,N,C] bf16 ----------------
__global__ __launch_bounds__(256) void av_kernel(
    const __bf16* __restrict__ attnb, const __bf16* __restrict__ vt,
    __bf16* __restrict__ Y) {
  __shared__ __attribute__((aligned(16))) __bf16 As[DH][104];   // [d][e]
  __shared__ __attribute__((aligned(16))) __bf16 Vs[DH][136];   // [e][n]
  const int bh = blockIdx.y;
  const int n0 = blockIdx.x * 128;
  const int tid = threadIdx.x, lane = tid & 31, wv = tid >> 5;
  const __bf16* ab = attnb + (size_t)bh * DH * DH;
  const __bf16* vb = vt + (size_t)bh * DH * NTOK;

  for (int g = tid; g < (DH * DH) / 8; g += 256) {
    const int r = g / 12, c = (g - r * 12) * 8;
    *(uint4*)&As[r][c] = *(const uint4*)&ab[r * DH + c];
  }
  for (int g = tid; g < (DH * 128) / 8; g += 256) {
    const int r = g >> 4, c = (g & 15) * 8;
    *(uint4*)&Vs[r][c] = *(const uint4*)&vb[(size_t)r * NTOK + n0 + c];
  }
  __syncthreads();

  v8f acc[6] = {};
#pragma unroll
  for (int kk = 0; kk < 3; ++kk) {
    const v16bf bfr = ld_frag_b_km(&Vs[kk * 32][wv * 16], 136);
#pragma unroll
    for (int mi = 0; mi < 6; ++mi) {
      const v16bf a = ld_frag_a_rm(&As[mi * 16][kk * 32], 104);
      acc[mi] = wmma_bf16(a, bfr, acc[mi]);
    }
  }

  const int b = bh >> 3, h = bh & (NH - 1);
  const int ng = n0 + wv * 16 + (lane & 15);
  const int hi = lane >> 4;
  __bf16* yrow = Y + ((size_t)(b * NTOK + ng)) * DIMC + h * DH;
#pragma unroll
  for (int mi = 0; mi < 6; ++mi) store_bf16x8(yrow + mi * 16 + hi * 8, acc[mi]);
}

// ---------------- Kernel 7: proj GEMM + bias + gamma1 + residual ----------------
__global__ __launch_bounds__(256) void proj_gemm_kernel(
    const __bf16* __restrict__ Yb, const __bf16* __restrict__ WpT,
    const float* __restrict__ X, const float* __restrict__ pb,
    const float* __restrict__ g1, float* __restrict__ Out) {
  __shared__ __attribute__((aligned(16))) __bf16 As[256][40];
  __shared__ __attribute__((aligned(16))) __bf16 Bs[32][136];
  const int tid = threadIdx.x, lane = tid & 31, wv = tid >> 5;
  const int m0 = blockIdx.y * 256, n0 = blockIdx.x * 128;
  const int wm = wv >> 1, wn = wv & 1;

  v8f acc[4][4] = {};
  for (int k0 = 0; k0 < DIMC; k0 += 32) {
#pragma unroll
    for (int it = 0; it < 4; ++it) {
      const int g = tid + it * 256;
      const int r = g >> 2, seg = g & 3;
      *(uint4*)&As[r][seg * 8] =
          *(const uint4*)&Yb[(size_t)(m0 + r) * DIMC + k0 + seg * 8];
    }
#pragma unroll
    for (int it = 0; it < 2; ++it) {
      const int g = tid + it * 256;
      const int r = g >> 4, seg = g & 15;
      *(uint4*)&Bs[r][seg * 8] =
          *(const uint4*)&WpT[(size_t)(k0 + r) * DIMC + n0 + seg * 8];
    }
    __syncthreads();
    v16bf af[4], bfv[4];
#pragma unroll
    for (int i = 0; i < 4; ++i) af[i] = ld_frag_a_rm(&As[wm * 64 + i * 16][0], 40);
#pragma unroll
    for (int j = 0; j < 4; ++j) bfv[j] = ld_frag_b_km(&Bs[0][wn * 64 + j * 16], 136);
#pragma unroll
    for (int i = 0; i < 4; ++i)
#pragma unroll
      for (int j = 0; j < 4; ++j) acc[i][j] = wmma_bf16(af[i], bfv[j], acc[i][j]);
    __syncthreads();
  }

  const int hi = lane >> 4, col = lane & 15;
#pragma unroll
  for (int i = 0; i < 4; ++i) {
    const int tbase = m0 + wm * 64 + i * 16 + hi * 8;
#pragma unroll
    for (int j = 0; j < 4; ++j) {
      const int o = n0 + wn * 64 + j * 16 + col;
      const float gg = g1[o], bb = pb[o];
#pragma unroll
      for (int r = 0; r < 8; ++r) {
        const size_t idx = (size_t)(tbase + r) * DIMC + o;
        Out[idx] = X[idx] + gg * (acc[i][j][r] + bb);
      }
    }
  }
}

// ---------------- Host launcher ----------------
extern "C" void kernel_launch(void* const* d_in, const int* in_sizes, int n_in,
                              void* d_out, int out_size, void* d_ws, size_t ws_size,
                              hipStream_t stream) {
  const float* x      = (const float*)d_in[0];
  const float* qkv_w  = (const float*)d_in[1];
  const float* proj_w = (const float*)d_in[2];
  const float* proj_b = (const float*)d_in[3];
  const float* ln_g   = (const float*)d_in[4];
  const float* ln_b   = (const float*)d_in[5];
  const float* temp   = (const float*)d_in[6];
  const float* gamma1 = (const float*)d_in[7];
  float* out = (float*)d_out;

  char* ws = (char*)d_ws;
  size_t off = 0;
  auto take = [&](size_t bytes) -> char* {
    char* p = ws + off;
    off += (bytes + 255) & ~(size_t)255;
    return p;
  };
  __bf16* WqT   = (__bf16*)take((size_t)3 * DIMC * DIMC * 2);
  __bf16* WpT   = (__bf16*)take((size_t)DIMC * DIMC * 2);
  __bf16* xn    = (__bf16*)take((size_t)MTOT * DIMC * 2);     // reused as Y
  __bf16* qt    = (__bf16*)take((size_t)BH * DH * NTOK * 2);
  __bf16* kt2   = (__bf16*)take((size_t)BH * DH * NTOK * 2);
  __bf16* vt    = (__bf16*)take((size_t)BH * DH * NTOK * 2);
  float*  qn    = (float*)take((size_t)BH * DH * 4);
  float*  kn    = (float*)take((size_t)BH * DH * 4);
  __bf16* attnb = (__bf16*)take((size_t)BH * DH * DH * 2);

  cvt_weights_kernel<<<1024, 256, 0, stream>>>(qkv_w, proj_w, WqT, WpT);
  ln_kernel<<<MTOT, 256, 0, stream>>>(x, ln_g, ln_b, xn);
  qkv_gemm_kernel<<<dim3(3 * DIMC / 128, MTOT / 256), 256, 0, stream>>>(
      xn, WqT, qt, kt2, vt);
  qnorm_kernel<<<BH * DH, 256, 0, stream>>>(qt, qn);
  knorm_kernel<<<BH, 128, 0, stream>>>(kt2, kn);
  attn_kernel<<<BH, 192, 0, stream>>>(qt, kt2, qn, kn, temp, attnb);
  av_kernel<<<dim3(NTOK / 128, BH), 256, 0, stream>>>(attnb, vt, xn /*as Y*/);
  proj_gemm_kernel<<<dim3(DIMC / 128, MTOT / 256), 256, 0, stream>>>(
      xn, WpT, x, proj_b, gamma1, out);
}
